// RNPGNNBase_31851477467847
// MI455X (gfx1250) — matrix operand order repaired and checked by reading, not compiled, not run on hardware
//
#include <hip/hip_runtime.h>

#define NNODES 640
#define HD 64
#define NP 80          // padded feature width: 64 x | 1 ones | 15 zero
#define BVS 648        // Bv column stride in halfs (bank-friendly, 16B aligned)
#define IST 104        // inp1 row stride (halfs), K padded 66 -> 96
#define HST 72         // hidden row stride (halfs), K = 64

typedef _Float16 h8   __attribute__((ext_vector_type(8)));
typedef _Float16 v16h __attribute__((ext_vector_type(16)));
typedef float    v8f  __attribute__((ext_vector_type(8)));

static __device__ __forceinline__ v16h cat16(h8 lo, h8 hi) {
  return __builtin_shufflevector(lo, hi, 0,1,2,3,4,5,6,7,8,9,10,11,12,13,14,15);
}
static __device__ __forceinline__ v8f wmma16(v16h a, v16h b, v8f c) {
  // D = A(16x32 f16) * B(32x16 f16) + C(16x16 f32)
  return __builtin_amdgcn_wmma_f32_16x16x32_f16(false, a, false, b, (short)0, c, false, false);
}
// B fragment from column-major Bv in LDS: lane = N column, K halfs 8h..8h+7 and 16+8h..16+8h+7
static __device__ __forceinline__ v16h loadB(const _Float16* __restrict__ Bv,
                                             int nt, int w0, int ln, int hi) {
  const _Float16* bp = Bv + (nt * 16 + ln) * BVS + w0 + 8 * hi;
  return cat16(*(const h8*)bp, *(const h8*)(bp + 16));
}

// Pre-swizzle a row-major [Kin x Nout] f32 weight matrix into WMMA B-fragment
// order: frag (kk, nt) is 32 lanes x 16 contiguous halfs; element e of lane l
// is W[kk*32 + 8*(l>>4) + (e&7) + 16*(e>>3)][nt*16 + (l&15)], zero-padded out of range.
static __device__ __forceinline__ void swz(const float* __restrict__ W, int Kin, int Nout,
                                           int NK, int NN, _Float16* __restrict__ dst,
                                           int gid, int nthr) {
  const int tot = NK * NN * 512;
  for (int i = gid; i < tot; i += nthr) {
    int frag = i >> 9;
    int wi   = i & 511;
    int lane = wi >> 4;
    int e    = wi & 15;
    int kk   = frag / NN;
    int nt   = frag - kk * NN;
    int K    = kk * 32 + ((lane >> 4) << 3) + (e & 7) + ((e >> 3) << 4);
    int Ncol = nt * 16 + (lane & 15);
    float v  = (K < Kin && Ncol < Nout) ? W[K * Nout + Ncol] : 0.0f;
    dst[i] = (_Float16)v;
  }
}

__global__ void k_init(const float* __restrict__ x,
                       const float* __restrict__ W1a, const float* __restrict__ W1b,
                       const float* __restrict__ W0a, const float* __restrict__ W0b,
                       _Float16* __restrict__ A, _Float16* __restrict__ Xext,
                       _Float16* __restrict__ sw1a, _Float16* __restrict__ sw1b,
                       _Float16* __restrict__ sw0a, _Float16* __restrict__ sw0b) {
  const int gid  = blockIdx.x * blockDim.x + threadIdx.x;
  const int nthr = gridDim.x * blockDim.x;
  for (int i = gid; i < NNODES * NNODES; i += nthr) A[i] = (_Float16)0.0f;
  // Xext column-major: Xext[d][w]; col 64 = ones so the main GEMM also yields cnt
  for (int i = gid; i < NP * NNODES; i += nthr) {
    int d = i / NNODES, w = i - d * NNODES;
    float v = (d < HD) ? x[w * HD + d] : (d == HD ? 1.0f : 0.0f);
    Xext[i] = (_Float16)v;
  }
  swz(W1a, HD + 2, HD,     3, 4, sw1a, gid, nthr);  // 66->64, K pad 96, N 64
  swz(W1b, HD,     HD + 1, 2, 5, sw1b, gid, nthr);  // 64->65, N pad 80
  swz(W0a, HD + 1, HD,     3, 4, sw0a, gid, nthr);  // 65->64, K pad 96
  swz(W0b, HD,     HD,     2, 4, sw0b, gid, nthr);  // 64->64
}

__global__ void k_edges(const int* __restrict__ ei, int E, _Float16* __restrict__ A) {
  int e = blockIdx.x * blockDim.x + threadIdx.x;
  if (e < E) {
    int s = ei[e], t = ei[E + e];
    if (s != t) A[s * NNODES + t] = (_Float16)1.0f;  // (1 - I) mask == skip diagonal
  }
}

// One workgroup per ego node v, 8 waves. Fully fused levels 1+aggregation.
// Each wave processes M=32 (two 16-row u-tiles) per pass so one B-fragment
// feeds two WMMAs (ds:wmma = 1:1). No explicit register double-buffering:
// the compiler clauses/hoists the ds loads itself; launch_bounds(256,1)
// matches the LDS-forced occupancy so nothing spills.
__global__ void __launch_bounds__(256, 1)
k_ego(const float* __restrict__ x,
      const float* __restrict__ b1a, const float* __restrict__ b1b,
      const _Float16* __restrict__ A, const _Float16* __restrict__ Xext,
      const _Float16* __restrict__ sw1a, const _Float16* __restrict__ sw1b,
      float* __restrict__ aggg) {
  extern __shared__ char smem[];
  _Float16* a_row = (_Float16*)smem;                                  // 640 h
  float*    agg   = (float*)(smem + 1280);                            // 80 f
  _Float16* Bv    = (_Float16*)(smem + 1600);                         // 80*BVS h
  _Float16* inpS  = (_Float16*)(smem + 1600 + NP * BVS * 2);          // 8 * 16*IST h
  _Float16* hS    = (_Float16*)(smem + 1600 + NP * BVS * 2 + 8 * 16 * IST * 2);

  const int v    = blockIdx.x;
  const int tid  = threadIdx.x;
  const int lane = tid & 31;
  const int wv   = tid >> 5;
  const int hi   = lane >> 4;
  const int ln   = lane & 15;

  // stage A[v,:] into LDS, zero agg
  for (int i = tid; i < NNODES / 8; i += 256)
    *(h8*)(a_row + i * 8) = *(const h8*)(A + (size_t)v * NNODES + i * 8);
  if (tid < NP) agg[tid] = 0.0f;
  __syncthreads();

  // Bv[d][w] = A[v,w] * Xext[d][w]  (column-major for B-fragment loads)
  for (int c = tid; c < NP * (NNODES / 8); c += 256) {
    int d  = c / (NNODES / 8);
    int wb = (c - d * (NNODES / 8)) * 8;
    h8 xv = *(const h8*)(Xext + d * NNODES + wb);
    h8 av = *(const h8*)(a_row + wb);
    *(h8*)(Bv + d * BVS + wb) = xv * av;
  }
  __syncthreads();

  _Float16* inp = inpS + wv * 16 * IST;
  _Float16* hh  = hS   + wv * 16 * HST;
  const v8f vz = {0.f, 0.f, 0.f, 0.f, 0.f, 0.f, 0.f, 0.f};

  for (int up = wv; up < NNODES / 32; up += 8) {
    const int u0 = up * 32;

    // ---- GEMM1: T_ext = A[u0:u0+32,:] @ Bv   (M=32, N=80, K=640) ----
    v8f acc[2][5];
#pragma unroll
    for (int s = 0; s < 2; ++s)
#pragma unroll
      for (int i = 0; i < 5; ++i) acc[s][i] = vz;

    for (int k = 0; k < NNODES / 32; ++k) {
      const int w0 = k * 32;
      const _Float16* ap0 = A + (size_t)(u0 + ln) * NNODES + w0 + 8 * hi;
      const _Float16* ap1 = ap0 + 16 * NNODES;
      __builtin_prefetch(ap0 + 32, 0, 1);
      __builtin_prefetch(ap1 + 32, 0, 1);
      v16h af0 = cat16(*(const h8*)ap0, *(const h8*)(ap0 + 16));
      v16h af1 = cat16(*(const h8*)ap1, *(const h8*)(ap1 + 16));
#pragma unroll
      for (int nt = 0; nt < 5; ++nt) {
        const v16h bf = loadB(Bv, nt, w0, ln, hi);
        acc[0][nt] = wmma16(af0, bf, acc[0][nt]);
        acc[1][nt] = wmma16(af1, bf, acc[1][nt]);
      }
    }

    // ---- per 16-row sub-tile: MLP level-1 + weighted aggregation ----
#pragma unroll 1
    for (int s = 0; s < 2; ++s) {
      const int u0s = u0 + s * 16;

      // compose inp1 = [x_u + T | 1+cnt | cnt | 0...] as f16 A-tile in LDS
      for (int i = lane; i < 16 * IST; i += 32) inp[i] = (_Float16)0.0f;
#pragma unroll
      for (int nt = 0; nt < 5; ++nt) {
        const int col = nt * 16 + ln;
#pragma unroll
        for (int r = 0; r < 8; ++r) {
          const int row = r + 8 * hi;
          float t = acc[s][nt][r];
          if (col < HD) {
            inp[row * IST + col] = (_Float16)(x[(u0s + row) * HD + col] + t);
          } else if (col == HD) {               // t == cnt here
            inp[row * IST + HD]     = (_Float16)(1.0f + t);
            inp[row * IST + HD + 1] = (_Float16)t;
          }
        }
      }

      // GEMM2: h = relu(inp1 @ W1a + b1a)   (M=16, N=64, K=96)
      v8f acc2[4];
#pragma unroll
      for (int i = 0; i < 4; ++i) acc2[i] = vz;
#pragma unroll
      for (int kk = 0; kk < 3; ++kk) {
        const _Float16* ap = inp + ln * IST + kk * 32 + 8 * hi;
        v16h af = cat16(*(const h8*)ap, *(const h8*)(ap + 16));
#pragma unroll
        for (int nt = 0; nt < 4; ++nt) {
          const v16h bf = *(const v16h*)(sw1a + (size_t)((kk * 4 + nt) * 32 + lane) * 16);
          acc2[nt] = wmma16(af, bf, acc2[nt]);
        }
      }
#pragma unroll
      for (int nt = 0; nt < 4; ++nt) {
        const int col = nt * 16 + ln;
        const float bb = b1a[col];
#pragma unroll
        for (int r = 0; r < 8; ++r) {
          float vv = fmaxf(acc2[nt][r] + bb, 0.0f);
          hh[(r + 8 * hi) * HST + col] = (_Float16)vv;
        }
      }

      // GEMM3: out1 = h @ W1b + b1b   (M=16, N=80, K=64)
      v8f acc3[5];
#pragma unroll
      for (int i = 0; i < 5; ++i) acc3[i] = vz;
#pragma unroll
      for (int kk = 0; kk < 2; ++kk) {
        const _Float16* ap = hh + ln * HST + kk * 32 + 8 * hi;
        v16h af = cat16(*(const h8*)ap, *(const h8*)(ap + 16));
#pragma unroll
        for (int nt = 0; nt < 5; ++nt) {
          const v16h bf = *(const v16h*)(sw1b + (size_t)((kk * 5 + nt) * 32 + lane) * 16);
          acc3[nt] = wmma16(af, bf, acc3[nt]);
        }
      }

      // agg[v,:] += sum_u A[v,u] * out1[u,:]  (out1 never hits memory)
      float ar[8];
#pragma unroll
      for (int r = 0; r < 8; ++r) ar[r] = (float)a_row[u0s + r + 8 * hi];
#pragma unroll
      for (int nt = 0; nt < 5; ++nt) {
        const int col = nt * 16 + ln;
        const float bb = (col < HD + 1) ? b1b[col] : 0.0f;
        float sum = 0.0f;
#pragma unroll
        for (int r = 0; r < 8; ++r) sum += ar[r] * (acc3[nt][r] + bb);
        atomicAdd(&agg[col], sum);   // ds_add_f32
      }
    }
  }

  __syncthreads();
  if (tid < NP) aggg[v * NP + tid] = agg[tid];
}

// Final node MLP: out = relu((hv+agg)@W0a + b0a)@W0b + b0b. One wave per 16 rows.
__global__ void __launch_bounds__(32, 1)
k_final(const float* __restrict__ x,
        const float* __restrict__ b0a, const float* __restrict__ b0b,
        const float* __restrict__ aggg,
        const _Float16* __restrict__ sw0a, const _Float16* __restrict__ sw0b,
        float* __restrict__ out) {
  extern __shared__ char smem[];
  _Float16* inp = (_Float16*)smem;                   // 16*IST halfs
  _Float16* hh  = (_Float16*)(smem + 16 * IST * 2);  // 16*HST halfs
  const int lane = threadIdx.x;
  const int hi = lane >> 4, ln = lane & 15;
  const int v0 = blockIdx.x * 16;
  const v8f vz = {0.f, 0.f, 0.f, 0.f, 0.f, 0.f, 0.f, 0.f};

  for (int i = lane; i < 16 * IST; i += 32) inp[i] = (_Float16)0.0f;
  for (int i = lane; i < 16 * (HD + 1); i += 32) {
    int row = i / (HD + 1), c = i - row * (HD + 1);
    float val = aggg[(v0 + row) * NP + c] + (c < HD ? x[(v0 + row) * HD + c] : 0.0f);
    inp[row * IST + c] = (_Float16)val;
  }
  __syncthreads();

  v8f acc[4];
#pragma unroll
  for (int i = 0; i < 4; ++i) acc[i] = vz;
#pragma unroll
  for (int kk = 0; kk < 3; ++kk) {
    const _Float16* ap = inp + ln * IST + kk * 32 + 8 * hi;
    v16h af = cat16(*(const h8*)ap, *(const h8*)(ap + 16));
#pragma unroll
    for (int nt = 0; nt < 4; ++nt) {
      const v16h bf = *(const v16h*)(sw0a + (size_t)((kk * 4 + nt) * 32 + lane) * 16);
      acc[nt] = wmma16(af, bf, acc[nt]);
    }
  }
#pragma unroll
  for (int nt = 0; nt < 4; ++nt) {
    const int col = nt * 16 + ln;
    const float bb = b0a[col];
#pragma unroll
    for (int r = 0; r < 8; ++r) {
      float vv = fmaxf(acc[nt][r] + bb, 0.0f);
      hh[(r + 8 * hi) * HST + col] = (_Float16)vv;
    }
  }
  __syncthreads();

  v8f acc2[4];
#pragma unroll
  for (int i = 0; i < 4; ++i) acc2[i] = vz;
#pragma unroll
  for (int kk = 0; kk < 2; ++kk) {
    const _Float16* ap = hh + ln * HST + kk * 32 + 8 * hi;
    v16h af = cat16(*(const h8*)ap, *(const h8*)(ap + 16));
#pragma unroll
    for (int nt = 0; nt < 4; ++nt) {
      const v16h bf = *(const v16h*)(sw0b + (size_t)((kk * 4 + nt) * 32 + lane) * 16);
      acc2[nt] = wmma16(af, bf, acc2[nt]);
    }
  }
#pragma unroll
  for (int nt = 0; nt < 4; ++nt) {
    const int col = nt * 16 + ln;
    const float bb = b0b[col];
#pragma unroll
    for (int r = 0; r < 8; ++r)
      out[(size_t)(v0 + r + 8 * hi) * HD + col] = acc2[nt][r] + bb;
  }
}

extern "C" void kernel_launch(void* const* d_in, const int* in_sizes, int n_in,
                              void* d_out, int out_size, void* d_ws, size_t ws_size,
                              hipStream_t stream) {
  (void)n_in; (void)out_size; (void)ws_size;
  const float* x   = (const float*)d_in[0];
  const float* W1a = (const float*)d_in[1];
  const float* b1a = (const float*)d_in[2];
  const float* W1b = (const float*)d_in[3];
  const float* b1b = (const float*)d_in[4];
  const float* W0a = (const float*)d_in[5];
  const float* b0a = (const float*)d_in[6];
  const float* W0b = (const float*)d_in[7];
  const float* b0b = (const float*)d_in[8];
  const int*   ei  = (const int*)d_in[9];
  const int E = in_sizes[9] / 2;

  char* ws = (char*)d_ws;
  _Float16* A    = (_Float16*)(ws);             // 819200 B
  _Float16* Xext = (_Float16*)(ws + 819200);    // 102400 B
  _Float16* sw1a = (_Float16*)(ws + 921600);    //  12288 B
  _Float16* sw1b = (_Float16*)(ws + 933888);    //  10240 B
  _Float16* sw0a = (_Float16*)(ws + 944128);    //  12288 B
  _Float16* sw0b = (_Float16*)(ws + 956416);    //   8192 B
  float*    aggg = (float*)(ws + 964608);       // 204800 B  (total ~1.17 MB)

  k_init<<<256, 256, 0, stream>>>(x, W1a, W1b, W0a, W0b, A, Xext, sw1a, sw1b, sw0a, sw0b);
  k_edges<<<(E + 255) / 256, 256, 0, stream>>>(ei, E, A);

  const size_t lds2 = 1600 + (size_t)NP * BVS * 2 + 8 * 16 * IST * 2 + 8 * 16 * HST * 2;
  k_ego<<<NNODES, 256, lds2, stream>>>(x, b1a, b1b, A, Xext, sw1a, sw1b, aggg);

  const size_t lds3 = (size_t)16 * IST * 2 + 16 * HST * 2;
  k_final<<<NNODES / 16, 32, lds3, stream>>>(x, b0a, b0b, aggg, sw0a, sw0b, (float*)d_out);
}
